// HesGNN_agg_28037546508938
// MI455X (gfx1250) — matrix operand order repaired and checked by reading, not compiled, not run on hardware
//
#include <hip/hip_runtime.h>
#include <hip/hip_bf16.h>

// ---------------------------------------------------------------------------
// HesGNN (SAGEConv x3 + encoder) for MI455X / gfx1250.
//
// Perf model: aggregation is the bottleneck (~410MB gather + ~410MB atomic
// writes per layer @ 23.3 TB/s); GEMMs (13 GFLOP total) ride the WMMA units.
//   - encoder + layer GEMMs: v_wmma_f32_16x16x32_f16, f32 accum
//   - aggregation: float4 gathers + unsafeAtomicAdd (global_atomic_add_f32,
//     no-return, STOREcnt-tracked fire-and-forget)
//   - mean division folded into layer-GEMM A-fragment load (row scale)
// ---------------------------------------------------------------------------

typedef __attribute__((ext_vector_type(16))) _Float16 v16h;
typedef __attribute__((ext_vector_type(8)))  float    v8f;

#define N_NODES 50000
#define N_EDGES 800000
#define IN_DIM  256
#define HID_DIM 128

// ---- WMMA fragment helpers (C/D per CDNA5 ISA §7.12.2; A/B k order contiguous) -----

// A-matrix 16x32 f16 (MxK), loaded from row-major f32 [.., lda], optional
// per-row scale. lane&15 = M; element e -> K = e + 16*(lane>=16).
__device__ __forceinline__ v16h load_a_frag(const float* __restrict__ A, int lda,
                                            int row0, int k0,
                                            const float* __restrict__ rowscale,
                                            int lane) {
  const int m     = lane & 15;
  const int khalf = (lane >> 4) << 4;
  const float s = rowscale ? rowscale[row0 + m] : 1.0f;
  const float* rp = A + (size_t)(row0 + m) * lda + k0;
  v16h a;
#pragma unroll
  for (int e = 0; e < 16; ++e) {
    const int k = e + khalf;
    a[e] = (_Float16)(rp[k] * s);
  }
  return a;
}

// B-matrix 32x16 f16 (KxN) representing W^T: B[k][n] = W[n0+n][k0+k].
// W is row-major [nrows, ldw]. lane&15 = N; element e -> K = e + 16*(lane>=16).
__device__ __forceinline__ v16h load_b_frag_wt(const float* __restrict__ W, int ldw,
                                               int n0, int k0, int lane) {
  const int n     = lane & 15;
  const int kbase = (lane >> 4) << 4;
  const float* wp = W + (size_t)(n0 + n) * ldw + k0 + kbase;
  v16h b;
#pragma unroll
  for (int e = 0; e < 16; ++e) b[e] = (_Float16)wp[e];
  return b;
}

// C/D 16x16 f32: VGPR r, lane L -> M = r + 8*(L>=16), N = L&15.
__device__ __forceinline__ void store_c_frag(float* __restrict__ C, int ldc,
                                             int row0, int n0, v8f c, int lane,
                                             const float* __restrict__ bias) {
  const int n     = lane & 15;
  const int mbase = (lane >> 4) << 3;
  const float bv  = bias ? bias[n0 + n] : 0.0f;
#pragma unroll
  for (int r = 0; r < 8; ++r) {
    C[(size_t)(row0 + mbase + r) * ldc + n0 + n] = c[r] + bv;
  }
}

__device__ __forceinline__ v8f wmma_f16(v16h a, v16h b, v8f c) {
  return __builtin_amdgcn_wmma_f32_16x16x32_f16(false, a, false, b,
                                                (short)0, c, false, false);
}

// ---- Kernel 1: in-degree ---------------------------------------------------
__global__ void hes_degree_kernel(const int* __restrict__ dst,
                                  float* __restrict__ deg) {
  const int e = blockIdx.x * blockDim.x + threadIdx.x;
  if (e < N_EDGES) unsafeAtomicAdd(&deg[dst[e]], 1.0f);
}

// deg -> 1/max(deg,1), in place
__global__ void hes_invdenom_kernel(float* __restrict__ deg) {
  const int i = blockIdx.x * blockDim.x + threadIdx.x;
  if (i < N_NODES) deg[i] = 1.0f / fmaxf(deg[i], 1.0f);
}

// ---- Kernel 2: encoder GEMM  h = x @ W_enc^T + b_enc ----------------------
// grid.x = 3125 (16 rows each), block = 256 (8 waves, one 16-col tile each)
__global__ void hes_encoder_gemm(const float* __restrict__ x,
                                 const float* __restrict__ W,
                                 const float* __restrict__ b,
                                 float* __restrict__ h) {
  const int lane = threadIdx.x & 31;
  const int wave = threadIdx.x >> 5;
  const int row0 = blockIdx.x << 4;
  const int n0   = wave << 4;
  v8f acc = {};
#pragma unroll
  for (int k0 = 0; k0 < IN_DIM; k0 += 32) {
    v16h a = load_a_frag(x, IN_DIM, row0, k0, nullptr, lane);
    v16h w = load_b_frag_wt(W, IN_DIM, n0, k0, lane);
    acc = wmma_f16(a, w, acc);
  }
  store_c_frag(h, HID_DIM, row0, n0, acc, lane, b);
}

// ---- Kernel 3: edge scatter-add  agg[dst] += h[src] -----------------------
// 32 threads per edge, float4 per thread (128 feats). Fire-and-forget f32
// atomics -> global_atomic_add_f32 (no return), saturates HBM write path.
__global__ void hes_scatter_kernel(const float* __restrict__ h,
                                   const int* __restrict__ src,
                                   const int* __restrict__ dst,
                                   float* __restrict__ agg) {
  const long long t = (long long)blockIdx.x * blockDim.x + threadIdx.x;
  const int edge = (int)(t >> 5);
  if (edge >= N_EDGES) return;
  const int f = (int)(t & 31) << 2;
  const int s = src[edge];
  const int d = dst[edge];
  const float4 v = *(const float4*)(h + (size_t)s * HID_DIM + f);
  float* o = agg + (size_t)d * HID_DIM + f;
  unsafeAtomicAdd(o + 0, v.x);
  unsafeAtomicAdd(o + 1, v.y);
  unsafeAtomicAdd(o + 2, v.z);
  unsafeAtomicAdd(o + 3, v.w);
}

// ---- Kernel 4: fused SAGE layer  hout = (agg*inv) @ Wl^T + bl + h @ Wr^T --
__global__ void hes_sage_gemm(const float* __restrict__ agg,
                              const float* __restrict__ inv_denom,
                              const float* __restrict__ h,
                              const float* __restrict__ Wl,
                              const float* __restrict__ bl,
                              const float* __restrict__ Wr,
                              float* __restrict__ hout) {
  const int lane = threadIdx.x & 31;
  const int wave = threadIdx.x >> 5;
  const int row0 = blockIdx.x << 4;
  const int n0   = wave << 4;
  v8f acc = {};
#pragma unroll
  for (int k0 = 0; k0 < HID_DIM; k0 += 32) {
    v16h a = load_a_frag(agg, HID_DIM, row0, k0, inv_denom, lane);  // mean fold
    v16h w = load_b_frag_wt(Wl, HID_DIM, n0, k0, lane);
    acc = wmma_f16(a, w, acc);
  }
#pragma unroll
  for (int k0 = 0; k0 < HID_DIM; k0 += 32) {
    v16h a = load_a_frag(h, HID_DIM, row0, k0, nullptr, lane);
    v16h w = load_b_frag_wt(Wr, HID_DIM, n0, k0, lane);
    acc = wmma_f16(a, w, acc);
  }
  store_c_frag(hout, HID_DIM, row0, n0, acc, lane, bl);
}

// ---------------------------------------------------------------------------
extern "C" void kernel_launch(void* const* d_in, const int* in_sizes, int n_in,
                              void* d_out, int out_size, void* d_ws, size_t ws_size,
                              hipStream_t stream) {
  const float* x      = (const float*)d_in[0];
  const int*   eidx   = (const int*)d_in[1];   // [2, N_EDGES] row-major int32
  const float* W_enc  = (const float*)d_in[2]; // [128, 256]
  const float* b_enc  = (const float*)d_in[3]; // [128]
  const float* W_l    = (const float*)d_in[4]; // [3, 128, 128]
  const float* b_l    = (const float*)d_in[5]; // [3, 128]
  const float* W_r    = (const float*)d_in[6]; // [3, 128, 128]
  float*       out    = (float*)d_out;

  const int* src = eidx;            // edge_index[0]
  const int* dst = eidx + N_EDGES;  // edge_index[1]

  // Workspace carve-up (~77 MB): hA | hB | agg | inv_denom
  const size_t NH = (size_t)N_NODES * HID_DIM;  // 6.4M floats
  float* hA  = (float*)d_ws;
  float* hB  = hA + NH;
  float* agg = hB + NH;
  float* inv = agg + NH;

  const int nTileBlocks = N_NODES / 16;  // 3125 (exact)

  // 1) degree -> inverse denom
  hipMemsetAsync(inv, 0, (size_t)N_NODES * sizeof(float), stream);
  hes_degree_kernel<<<(N_EDGES + 255) / 256, 256, 0, stream>>>(dst, inv);
  hes_invdenom_kernel<<<(N_NODES + 255) / 256, 256, 0, stream>>>(inv);

  // 2) encoder
  hes_encoder_gemm<<<nTileBlocks, 256, 0, stream>>>(x, W_enc, b_enc, hA);

  // 3) layers (ping-pong hA/hB, last layer writes d_out)
  const long long scatterThreads = (long long)N_EDGES * 32;
  const int scatterBlocks = (int)((scatterThreads + 255) / 256);
  float* cur = hA;
  float* nxt = hB;
  for (int i = 0; i < 3; ++i) {
    float* outbuf = (i == 2) ? out : nxt;
    hipMemsetAsync(agg, 0, NH * sizeof(float), stream);
    hes_scatter_kernel<<<scatterBlocks, 256, 0, stream>>>(cur, src, dst, agg);
    hes_sage_gemm<<<nTileBlocks, 256, 0, stream>>>(
        agg, inv, cur, W_l + (size_t)i * HID_DIM * HID_DIM, b_l + (size_t)i * HID_DIM,
        W_r + (size_t)i * HID_DIM * HID_DIM, outbuf);
    float* t = cur; cur = outbuf == out ? cur : outbuf; nxt = t;
  }
}
